// PointNet2_1915555414493
// MI455X (gfx1250) — compile-verified, matrix operand order checked by reading, not run on hardware
//
#include <hip/hip_runtime.h>
#include <math.h>

namespace {
constexpr int   B_   = 8;
constexpr int   N_   = 4096;
constexpr int   K_   = 64;
constexpr int   M1_  = 1024;
constexpr int   M2_  = 256;
constexpr float R1SQ = 0.15f * 0.15f;
constexpr float R2SQ = 0.25f * 0.25f;
constexpr float BNINV = 0.9999950000374997f;   // 1/sqrt(1+1e-5)
}

typedef _Float16 v16h __attribute__((ext_vector_type(16)));
typedef _Float16 v8h  __attribute__((ext_vector_type(8)));
typedef float    v8f  __attribute__((ext_vector_type(8)));

// ---------------------------------------------------------------- transpose
__global__ void k_prep(const float* __restrict__ cloud, const float* __restrict__ xyz,
                       float* __restrict__ pos0, float* __restrict__ feats) {
  int i = blockIdx.x * 256 + threadIdx.x;
  if (i >= B_ * N_) return;
  int b = i / N_, p = i % N_;
  for (int d = 0; d < 3; ++d) pos0[(size_t)i * 3 + d] = xyz[((size_t)b * 3 + d) * N_ + p];
  for (int c = 0; c < 7; ++c) feats[(size_t)i * 7 + c] = cloud[((size_t)b * 9 + c + 2) * N_ + p];
}

// ---------------------------------------------------------------- FPS
template <int NP, int M>
__global__ void k_fps(const float* __restrict__ pos, int* __restrict__ idx) {
  __shared__ float mind[NP];
  __shared__ float rv[256];
  __shared__ int   ri[256];
  __shared__ float cx, cy, cz;
  const int b = blockIdx.x;
  const int t = threadIdx.x;
  const float* pb = pos + (size_t)b * NP * 3;
  float x0 = pb[0], y0 = pb[1], z0 = pb[2];
  for (int j = t; j < NP; j += 256) {
    float dx = pb[j * 3] - x0, dy = pb[j * 3 + 1] - y0, dz = pb[j * 3 + 2] - z0;
    mind[j] = dx * dx + dy * dy + dz * dz;
  }
  if (t == 0) idx[b * M] = 0;
  __syncthreads();
  for (int it = 1; it < M; ++it) {
    float bv = -1.0f; int bi = 0x7fffffff;
    for (int j = t; j < NP; j += 256) {
      float v = mind[j];
      if (v > bv || (v == bv && j < bi)) { bv = v; bi = j; }
    }
    rv[t] = bv; ri[t] = bi;
    __syncthreads();
    for (int s = 128; s > 0; s >>= 1) {
      if (t < s) {
        float v2 = rv[t + s]; int i2 = ri[t + s];
        if (v2 > rv[t] || (v2 == rv[t] && i2 < ri[t])) { rv[t] = v2; ri[t] = i2; }
      }
      __syncthreads();
    }
    if (t == 0) {
      int n = ri[0];
      idx[b * M + it] = n;
      cx = pb[n * 3]; cy = pb[n * 3 + 1]; cz = pb[n * 3 + 2];
    }
    __syncthreads();
    float sx = cx, sy = cy, sz = cz;
    for (int j = t; j < NP; j += 256) {
      float dx = pb[j * 3] - sx, dy = pb[j * 3 + 1] - sy, dz = pb[j * 3 + 2] - sz;
      float d = dx * dx + dy * dy + dz * dz;
      if (d < mind[j]) mind[j] = d;
    }
    __syncthreads();
  }
}

// ---------------------------------------------------------------- gather centers
__global__ void k_gather3(const float* __restrict__ pos, const int* __restrict__ idx,
                          float* __restrict__ posc, int M, int NP) {
  int i = blockIdx.x * 256 + threadIdx.x;
  if (i >= B_ * M) return;
  int b = i / M;
  int j = idx[i];
  for (int d = 0; d < 3; ++d) posc[(size_t)i * 3 + d] = pos[((size_t)b * NP + j) * 3 + d];
}

// ---------------------------------------------------------------- ball query (K smallest, top_k tie semantics)
template <int NP>
__global__ void k_ballq(const float* __restrict__ pos, const float* __restrict__ posc,
                        int M, int* __restrict__ nbr, float* __restrict__ d2o) {
  __shared__ float d2s[NP];
  __shared__ float rv[128];
  __shared__ int   ri[128];
  const int cidx = blockIdx.x;           // b*M + m
  const int b = cidx / M;
  const int t = threadIdx.x;
  const float* pb = pos + (size_t)b * NP * 3;
  const float cx = posc[(size_t)cidx * 3], cy = posc[(size_t)cidx * 3 + 1], cz = posc[(size_t)cidx * 3 + 2];
  for (int j = t; j < NP; j += 128) {
    float dx = pb[j * 3] - cx, dy = pb[j * 3 + 1] - cy, dz = pb[j * 3 + 2] - cz;
    d2s[j] = dx * dx + dy * dy + dz * dz;
  }
  __syncthreads();
  for (int kk = 0; kk < K_; ++kk) {
    float bv = 3.4e38f; int bi = 0x7fffffff;
    for (int j = t; j < NP; j += 128) {
      float v = d2s[j];
      if (v < bv || (v == bv && j < bi)) { bv = v; bi = j; }
    }
    rv[t] = bv; ri[t] = bi;
    __syncthreads();
    for (int s = 64; s > 0; s >>= 1) {
      if (t < s) {
        float v2 = rv[t + s]; int i2 = ri[t + s];
        if (v2 < rv[t] || (v2 == rv[t] && i2 < ri[t])) { rv[t] = v2; ri[t] = i2; }
      }
      __syncthreads();
    }
    if (t == 0) {
      nbr[(size_t)cidx * K_ + kk] = ri[0];
      d2o[(size_t)cidx * K_ + kk] = rv[0];
      d2s[ri[0]] = 3.4e38f;
    }
    __syncthreads();
  }
}

// ---------------------------------------------------------------- PointConv SA1: [10->16->16], masked max over K
__global__ void k_pc_sa1(const float* __restrict__ feats, const float* __restrict__ pos,
                         const float* __restrict__ posc, const int* __restrict__ nbr,
                         const float* __restrict__ d2,
                         const float* W0, const float* b0, const float* g0, const float* be0,
                         const float* W1, const float* b1, const float* g1, const float* be1,
                         float* __restrict__ x1) {
  __shared__ float hs[64 * 16];
  const int cidx = blockIdx.x;           // b*M1 + m
  const int b = cidx / M1_;
  const int t = threadIdx.x;             // neighbor slot
  const int j = nbr[(size_t)cidx * K_ + t];
  const bool valid = d2[(size_t)cidx * K_ + t] <= R1SQ;
  float in[10];
  const float* fp = feats + ((size_t)b * N_ + j) * 7;
  for (int c = 0; c < 7; ++c) in[c] = fp[c];
  const float* pj = pos + ((size_t)b * N_ + j) * 3;
  in[7] = pj[0] - posc[(size_t)cidx * 3];
  in[8] = pj[1] - posc[(size_t)cidx * 3 + 1];
  in[9] = pj[2] - posc[(size_t)cidx * 3 + 2];
  float h1[16];
  for (int o = 0; o < 16; ++o) {
    float s = b0[o];
    for (int c = 0; c < 10; ++c) s += in[c] * W0[c * 16 + o];
    s = fmaxf(s, 0.0f);
    h1[o] = s * (g0[o] * BNINV) + be0[o];
  }
  for (int o = 0; o < 16; ++o) {
    float s = b1[o];
    for (int c = 0; c < 16; ++c) s += h1[c] * W1[c * 16 + o];
    s = fmaxf(s, 0.0f);
    hs[t * 16 + o] = valid ? (s * (g1[o] * BNINV) + be1[o]) : -INFINITY;
  }
  __syncthreads();
  if (t < 16) {
    float m = -INFINITY;
    for (int n = 0; n < 64; ++n) m = fmaxf(m, hs[n * 16 + t]);
    x1[(size_t)cidx * 16 + t] = m;
  }
}

// ---------------------------------------------------------------- PointConv SA2: [19->32], masked max over K
__global__ void k_pc_sa2(const float* __restrict__ x1, const float* __restrict__ pos1,
                         const float* __restrict__ posc, const int* __restrict__ nbr,
                         const float* __restrict__ d2,
                         const float* W, const float* bb, const float* g, const float* be,
                         float* __restrict__ x2) {
  __shared__ float hs[64 * 32];
  const int cidx = blockIdx.x;           // b*M2 + m
  const int b = cidx / M2_;
  const int t = threadIdx.x;
  const int j = nbr[(size_t)cidx * K_ + t];
  const bool valid = d2[(size_t)cidx * K_ + t] <= R2SQ;
  float in[19];
  const float* xp = x1 + ((size_t)b * M1_ + j) * 16;
  for (int c = 0; c < 16; ++c) in[c] = xp[c];
  const float* pj = pos1 + ((size_t)b * M1_ + j) * 3;
  in[16] = pj[0] - posc[(size_t)cidx * 3];
  in[17] = pj[1] - posc[(size_t)cidx * 3 + 1];
  in[18] = pj[2] - posc[(size_t)cidx * 3 + 2];
  for (int o = 0; o < 32; ++o) {
    float s = bb[o];
    for (int c = 0; c < 19; ++c) s += in[c] * W[c * 32 + o];
    s = fmaxf(s, 0.0f);
    hs[t * 32 + o] = valid ? (s * (g[o] * BNINV) + be[o]) : -INFINITY;
  }
  __syncthreads();
  if (t < 32) {
    float m = -INFINITY;
    for (int n = 0; n < 64; ++n) m = fmaxf(m, hs[n * 32 + t]);
    x2[(size_t)cidx * 32 + t] = m;
  }
}

// ---------------------------------------------------------------- SA3 per-point MLP [35->64]
__global__ void k_sa3(const float* __restrict__ x2, const float* __restrict__ pos2,
                      const float* W, const float* bb, const float* g, const float* be,
                      float* __restrict__ h3) {
  int i = blockIdx.x * 256 + threadIdx.x;
  if (i >= B_ * M2_) return;
  float in[35];
  for (int c = 0; c < 32; ++c) in[c] = x2[(size_t)i * 32 + c];
  for (int d = 0; d < 3; ++d) in[32 + d] = pos2[(size_t)i * 3 + d];
  for (int o = 0; o < 64; ++o) {
    float s = bb[o];
    for (int c = 0; c < 35; ++c) s += in[c] * W[c * 64 + o];
    s = fmaxf(s, 0.0f);
    h3[(size_t)i * 64 + o] = s * (g[o] * BNINV) + be[o];
  }
}

__global__ void k_sa3max(const float* __restrict__ h3, float* __restrict__ x3) {
  int b = blockIdx.x, o = threadIdx.x;   // 64 threads
  float m = -INFINITY;
  for (int j = 0; j < M2_; ++j) m = fmaxf(m, h3[((size_t)b * M2_ + j) * 64 + o]);
  x3[(size_t)b * 64 + o] = m;
}

// ---------------------------------------------------------------- FP3: broadcast x3 (k=1 interp), concat x2, [96->64]
__global__ void k_fp3(const float* __restrict__ x3, const float* __restrict__ x2,
                      const float* W, const float* bb, const float* g, const float* be,
                      float* __restrict__ xfp3) {
  int i = blockIdx.x * 256 + threadIdx.x;
  if (i >= B_ * M2_) return;
  int b = i / M2_;
  float in[96];
  for (int c = 0; c < 64; ++c) in[c] = x3[(size_t)b * 64 + c];
  for (int c = 0; c < 32; ++c) in[64 + c] = x2[(size_t)i * 32 + c];
  for (int o = 0; o < 64; ++o) {
    float s = bb[o];
    for (int c = 0; c < 96; ++c) s += in[c] * W[c * 64 + o];
    s = fmaxf(s, 0.0f);
    xfp3[(size_t)i * 64 + o] = s * (g[o] * BNINV) + be[o];
  }
}

// ---------------------------------------------------------------- FP2: 3-NN interp (256 src), concat x1, [80->34]
__global__ void k_fp2(const float* __restrict__ xfp3, const float* __restrict__ pos2,
                      const float* __restrict__ pos1, const float* __restrict__ x1,
                      const float* W, const float* bb, const float* g, const float* be,
                      float* __restrict__ xfp2) {
  int i = blockIdx.x * 256 + threadIdx.x;
  if (i >= B_ * M1_) return;
  int b = i / M1_;
  float px = pos1[(size_t)i * 3], py = pos1[(size_t)i * 3 + 1], pz = pos1[(size_t)i * 3 + 2];
  float bd0 = 3.4e38f, bd1 = 3.4e38f, bd2 = 3.4e38f;
  int   bi0 = 0, bi1 = 0, bi2 = 0;
  const float* sp = pos2 + (size_t)b * M2_ * 3;
  for (int s = 0; s < M2_; ++s) {
    float dx = px - sp[s * 3], dy = py - sp[s * 3 + 1], dz = pz - sp[s * 3 + 2];
    float d = dx * dx + dy * dy + dz * dz;
    if (d < bd0)      { bd2 = bd1; bi2 = bi1; bd1 = bd0; bi1 = bi0; bd0 = d; bi0 = s; }
    else if (d < bd1) { bd2 = bd1; bi2 = bi1; bd1 = d; bi1 = s; }
    else if (d < bd2) { bd2 = d; bi2 = s; }
  }
  float w0 = 1.0f / fmaxf(bd0, 1e-16f);
  float w1 = 1.0f / fmaxf(bd1, 1e-16f);
  float w2 = 1.0f / fmaxf(bd2, 1e-16f);
  float inv = 1.0f / (w0 + w1 + w2);
  const float* f0 = xfp3 + ((size_t)b * M2_ + bi0) * 64;
  const float* f1 = xfp3 + ((size_t)b * M2_ + bi1) * 64;
  const float* f2 = xfp3 + ((size_t)b * M2_ + bi2) * 64;
  float in[80];
  for (int c = 0; c < 64; ++c) in[c] = (w0 * f0[c] + w1 * f1[c] + w2 * f2[c]) * inv;
  for (int c = 0; c < 16; ++c) in[64 + c] = x1[(size_t)i * 16 + c];
  for (int o = 0; o < 34; ++o) {
    float s = bb[o];
    for (int c = 0; c < 80; ++c) s += in[c] * W[c * 34 + o];
    s = fmaxf(s, 0.0f);
    xfp2[(size_t)i * 34 + o] = s * (g[o] * BNINV) + be[o];
  }
}

// ---------------------------------------------------------------- FP1 interpolation -> f16 A rows [B*N, 64]
__global__ void k_fp1_interp(const float* __restrict__ xfp2, const float* __restrict__ pos1,
                             const float* __restrict__ pos0, const float* __restrict__ feats,
                             _Float16* __restrict__ a16) {
  __shared__ float ps[M1_ * 3];
  const int blocksPerBatch = N_ / 256;
  int b = blockIdx.x / blocksPerBatch;
  int blk = blockIdx.x % blocksPerBatch;
  for (int j = threadIdx.x; j < M1_ * 3; j += 256) ps[j] = pos1[(size_t)b * M1_ * 3 + j];
  __syncthreads();
  int p = blk * 256 + threadIdx.x;
  int gi = b * N_ + p;
  __builtin_prefetch(feats + (size_t)gi * 7, 0, 0);
  float px = pos0[(size_t)gi * 3], py = pos0[(size_t)gi * 3 + 1], pz = pos0[(size_t)gi * 3 + 2];
  float bd0 = 3.4e38f, bd1 = 3.4e38f, bd2 = 3.4e38f;
  int   bi0 = 0, bi1 = 0, bi2 = 0;
  for (int s = 0; s < M1_; ++s) {
    float dx = px - ps[s * 3], dy = py - ps[s * 3 + 1], dz = pz - ps[s * 3 + 2];
    float d = dx * dx + dy * dy + dz * dz;
    if (d < bd0)      { bd2 = bd1; bi2 = bi1; bd1 = bd0; bi1 = bi0; bd0 = d; bi0 = s; }
    else if (d < bd1) { bd2 = bd1; bi2 = bi1; bd1 = d; bi1 = s; }
    else if (d < bd2) { bd2 = d; bi2 = s; }
  }
  float w0 = 1.0f / fmaxf(bd0, 1e-16f);
  float w1 = 1.0f / fmaxf(bd1, 1e-16f);
  float w2 = 1.0f / fmaxf(bd2, 1e-16f);
  float inv = 1.0f / (w0 + w1 + w2);
  const float* f0 = xfp2 + ((size_t)b * M1_ + bi0) * 34;
  const float* f1 = xfp2 + ((size_t)b * M1_ + bi1) * 34;
  const float* f2 = xfp2 + ((size_t)b * M1_ + bi2) * 34;
  _Float16* row = a16 + (size_t)gi * 64;
  for (int c = 0; c < 34; ++c)
    row[c] = (_Float16)((w0 * f0[c] + w1 * f1[c] + w2 * f2[c]) * inv);
  for (int c = 0; c < 7; ++c) row[34 + c] = (_Float16)feats[(size_t)gi * 7 + c];
  for (int c = 41; c < 64; ++c) row[c] = (_Float16)0.0f;
}

// ---------------------------------------------------------------- FP1 weights -> f16, pre-swizzled into per-lane
// WMMA B-fragment order: Wh[((nt*2 + ks)*32 + lane)*16 + h]
// where col = nt*16 + (lane&15), k = (h&7) + ((h>>3)<<4) + (lane>>4)*8 + ks*32
__global__ void k_wprep(const float* __restrict__ W, _Float16* __restrict__ Wh) {
  int i = blockIdx.x * 256 + threadIdx.x;  // 3*2*32*16 = 3072 entries
  if (i >= 3 * 2 * 32 * 16) return;
  int h    = i & 15;
  int lane = (i >> 4) & 31;
  int ks   = (i >> 9) & 1;
  int nt   = i >> 10;
  int col = nt * 16 + (lane & 15);
  int k = (h & 7) + ((h >> 3) << 4) + (lane >> 4) * 8 + ks * 32;
  Wh[i] = (k < 41 && col < 34) ? (_Float16)W[k * 34 + col] : (_Float16)0.0f;
}

// ---------------------------------------------------------------- FP1 GEMM via WMMA: [16 pts x 64] x [64 x 48]
__global__ void k_fp1_wmma(const _Float16* __restrict__ A, const _Float16* __restrict__ Wh,
                           const float* __restrict__ bias, const float* __restrict__ g,
                           const float* __restrict__ be, float* __restrict__ out) {
  const int wave = threadIdx.x >> 5;
  const int lane = threadIdx.x & 31;
  const int tile = blockIdx.x * 8 + wave;            // 2048 tiles of 16 points
  const int rowbase = tile * 16;
  const int r  = lane & 15;
  const int hi = lane >> 4;
  // A fragments, two k-steps of K=32 (documented 16-bit A layout).
  // Per lane & k-step: two contiguous 8-half (16B) chunks -> b128 loads.
  const v8h* ar = (const v8h*)(A + (size_t)(rowbase + r) * 64);   // 8 chunks per row
  v8h lo0 = ar[hi];          // k = hi*8 + {0..7}
  v8h up0 = ar[hi + 2];      // k = hi*8 + 16 + {0..7}
  v8h lo1 = ar[hi + 4];      // +32
  v8h up1 = ar[hi + 6];
  v16h a0 = __builtin_shufflevector(lo0, up0, 0, 1, 2, 3, 4, 5, 6, 7, 8, 9, 10, 11, 12, 13, 14, 15);
  v16h a1 = __builtin_shufflevector(lo1, up1, 0, 1, 2, 3, 4, 5, 6, 7, 8, 9, 10, 11, 12, 13, 14, 15);
  // B fragments: pre-swizzled, one v16h (32B) vector load each.
  const v16h* wf = (const v16h*)Wh;                  // [nt][ks][lane]
  v8f acc[3];
  for (int nt = 0; nt < 3; ++nt) {
    v16h bf0 = wf[(nt * 2 + 0) * 32 + lane];
    v16h bf1 = wf[(nt * 2 + 1) * 32 + lane];
    v8f c = {};
    c = __builtin_amdgcn_wmma_f32_16x16x32_f16(false, a0, false, bf0, (short)0, c, false, false);
    c = __builtin_amdgcn_wmma_f32_16x16x32_f16(false, a1, false, bf1, (short)0, c, false, false);
    acc[nt] = c;
  }
  for (int nt = 0; nt < 3; ++nt) {
    int col = nt * 16 + r;
    if (col < 34) {
      float gs = g[col] * BNINV, bb = bias[col], bo = be[col];
      for (int v = 0; v < 8; ++v) {
        int row = rowbase + v + hi * 8;
        float val = acc[nt][v] + bb;
        val = fmaxf(val, 0.0f);
        out[(size_t)row * 34 + col] = val * gs + bo;
      }
    }
  }
}

// ---------------------------------------------------------------- head: lin1+relu, lin2, softmax/sigmoid
__global__ void k_head(const float* __restrict__ x,
                       const float* __restrict__ W1, const float* __restrict__ b1,
                       const float* __restrict__ W2, const float* __restrict__ b2,
                       float* __restrict__ out) {
  int p = blockIdx.x * 256 + threadIdx.x;
  if (p >= B_ * N_) return;
  const float* xp = x + (size_t)p * 34;
  float h[16];
  for (int o = 0; o < 16; ++o) {
    float s = b1[o];
    for (int c = 0; c < 34; ++c) s += xp[c] * W1[c * 16 + o];
    h[o] = fmaxf(s, 0.0f);
  }
  float sc[5];
  for (int o = 0; o < 5; ++o) {
    float s = b2[o];
    for (int c = 0; c < 16; ++c) s += h[c] * W2[c * 5 + o];
    sc[o] = s;
  }
  float m = fmaxf(fmaxf(sc[0], sc[1]), fmaxf(sc[2], sc[3]));
  float e[4], es = 0.0f;
  for (int c = 0; c < 4; ++c) { e[c] = expf(sc[c] - m); es += e[c]; }
  float dens = 1.0f / (1.0f + expf(-sc[4]));
  for (int c = 0; c < 4; ++c) {
    float pr = e[c] / es;
    out[(size_t)p * 4 + c] = pr * dens;                         // cov
    out[(size_t)(B_ * N_) * 4 + (size_t)p * 4 + c] = pr;        // proba
  }
}

// ================================================================ host
extern "C" void kernel_launch(void* const* d_in, const int* in_sizes, int n_in,
                              void* d_out, int out_size, void* d_ws, size_t ws_size,
                              hipStream_t stream) {
  const float* cloud = (const float*)d_in[0];
  const float* xyz   = (const float*)d_in[1];

  // Param flattening order: insertion (sa1 first) vs sorted keys (fp1 first).
  int sa1, sa2, sa3, fp3, fp2, fp1, l1, l2;
  if (n_in > 2 && in_sizes[2] == 160) {          // sa1 layer0 W [10,16]
    sa1 = 2; sa2 = 10; sa3 = 14; fp3 = 18; fp2 = 22; fp1 = 26; l1 = 30; l2 = 32;
  } else {                                       // sorted: fp1 W [41,34] = 1394
    fp1 = 2; fp2 = 6; fp3 = 10; l1 = 14; l2 = 16; sa1 = 18; sa2 = 26; sa3 = 30;
  }
  auto F = [&](int i) { return (const float*)d_in[i]; };

  // workspace bump allocator
  char* wp = (char*)d_ws;
  auto alloc = [&](size_t bytes) -> void* {
    void* p = wp;
    wp += (bytes + 255) & ~(size_t)255;
    return p;
  };
  float* pos0  = (float*)alloc((size_t)B_ * N_ * 3 * 4);
  float* feats = (float*)alloc((size_t)B_ * N_ * 7 * 4);
  int*   idx1  = (int*)  alloc((size_t)B_ * M1_ * 4);
  float* pos1  = (float*)alloc((size_t)B_ * M1_ * 3 * 4);
  int*   nbr1  = (int*)  alloc((size_t)B_ * M1_ * K_ * 4);
  float* d21   = (float*)alloc((size_t)B_ * M1_ * K_ * 4);
  float* x1    = (float*)alloc((size_t)B_ * M1_ * 16 * 4);
  int*   idx2  = (int*)  alloc((size_t)B_ * M2_ * 4);
  float* pos2  = (float*)alloc((size_t)B_ * M2_ * 3 * 4);
  int*   nbr2  = (int*)  alloc((size_t)B_ * M2_ * K_ * 4);
  float* d22   = (float*)alloc((size_t)B_ * M2_ * K_ * 4);
  float* x2    = (float*)alloc((size_t)B_ * M2_ * 32 * 4);
  float* h3    = (float*)alloc((size_t)B_ * M2_ * 64 * 4);
  float* x3    = (float*)alloc((size_t)B_ * 64 * 4);
  float* xfp3  = (float*)alloc((size_t)B_ * M2_ * 64 * 4);
  float* xfp2  = (float*)alloc((size_t)B_ * M1_ * 34 * 4);
  _Float16* a16 = (_Float16*)alloc((size_t)B_ * N_ * 64 * 2);
  _Float16* wh  = (_Float16*)alloc((size_t)3 * 2 * 32 * 16 * 2);
  float* xfp1  = (float*)alloc((size_t)B_ * N_ * 34 * 4);

  // 1) transpose inputs
  k_prep<<<(B_ * N_ + 255) / 256, 256, 0, stream>>>(cloud, xyz, pos0, feats);
  // 2) SA1: FPS + gather + ball query + pointconv
  k_fps<N_, M1_><<<B_, 256, 0, stream>>>(pos0, idx1);
  k_gather3<<<(B_ * M1_ + 255) / 256, 256, 0, stream>>>(pos0, idx1, pos1, M1_, N_);
  k_ballq<N_><<<B_ * M1_, 128, 0, stream>>>(pos0, pos1, M1_, nbr1, d21);
  k_pc_sa1<<<B_ * M1_, 64, 0, stream>>>(feats, pos0, pos1, nbr1, d21,
                                        F(sa1 + 0), F(sa1 + 1), F(sa1 + 2), F(sa1 + 3),
                                        F(sa1 + 4), F(sa1 + 5), F(sa1 + 6), F(sa1 + 7), x1);
  // 3) SA2
  k_fps<M1_, M2_><<<B_, 256, 0, stream>>>(pos1, idx2);
  k_gather3<<<(B_ * M2_ + 255) / 256, 256, 0, stream>>>(pos1, idx2, pos2, M2_, M1_);
  k_ballq<M1_><<<B_ * M2_, 128, 0, stream>>>(pos1, pos2, M2_, nbr2, d22);
  k_pc_sa2<<<B_ * M2_, 64, 0, stream>>>(x1, pos1, pos2, nbr2, d22,
                                        F(sa2 + 0), F(sa2 + 1), F(sa2 + 2), F(sa2 + 3), x2);
  // 4) SA3 global
  k_sa3<<<(B_ * M2_ + 255) / 256, 256, 0, stream>>>(x2, pos2,
                                                    F(sa3 + 0), F(sa3 + 1), F(sa3 + 2), F(sa3 + 3), h3);
  k_sa3max<<<B_, 64, 0, stream>>>(h3, x3);
  // 5) FP3 / FP2
  k_fp3<<<(B_ * M2_ + 255) / 256, 256, 0, stream>>>(x3, x2,
                                                    F(fp3 + 0), F(fp3 + 1), F(fp3 + 2), F(fp3 + 3), xfp3);
  k_fp2<<<(B_ * M1_ + 255) / 256, 256, 0, stream>>>(xfp3, pos2, pos1, x1,
                                                    F(fp2 + 0), F(fp2 + 1), F(fp2 + 2), F(fp2 + 3), xfp2);
  // 6) FP1 interp -> f16 A rows; swizzle weights; WMMA GEMM + BN epilogue
  k_fp1_interp<<<B_ * (N_ / 256), 256, 0, stream>>>(xfp2, pos1, pos0, feats, a16);
  k_wprep<<<(3 * 2 * 32 * 16 + 255) / 256, 256, 0, stream>>>(F(fp1 + 0), wh);
  k_fp1_wmma<<<(B_ * N_ / 16) / 8, 256, 0, stream>>>(a16, wh, F(fp1 + 1), F(fp1 + 2), F(fp1 + 3), xfp1);
  // 7) head
  k_head<<<(B_ * N_ + 255) / 256, 256, 0, stream>>>(xfp1, F(l1 + 0), F(l1 + 1),
                                                    F(l2 + 0), F(l2 + 1), (float*)d_out);
}
